// EncoderNetwork_27049704030335
// MI455X (gfx1250) — matrix operand order repaired and checked by reading, not compiled, not run on hardware
//
#include <hip/hip_runtime.h>

typedef unsigned short bf16_t;
typedef __attribute__((ext_vector_type(16))) __bf16          v16bf;
typedef __attribute__((ext_vector_type(16))) unsigned short  v16u;
typedef __attribute__((ext_vector_type(8)))  float           v8f;

#define D_MODEL 512
#define N_HEAD 8
#define HEAD_SIZE 64
#define SEQ 512
#define BATCH 8
#define ROWS (BATCH * SEQ)   /* 4096 */
#define FF_DIM 2048
#define NORM_EPS 1e-6f

// hardware bf16 convert (single v_cvt op)
__device__ __forceinline__ unsigned short f2bf(float x) {
  union { __bf16 b; unsigned short u; } c;
  c.b = (__bf16)x;
  return c.u;
}
__device__ __forceinline__ v16bf as_bf(v16u u) {
  union { v16u u; v16bf b; } c; c.u = u; return c.b;
}

// ---------------- WMMA 16x16x32 bf16 fragment loaders (wave32) ----------------
// A: 16x32 MxK. lane L: row M=L%16; elems 0..15 = K kb..kb+15, kb=(L/16)*16
__device__ __forceinline__ v16bf load_a_bf16(const bf16_t* S, int ld) {
  int lane = threadIdx.x & 31;
  const bf16_t* p = S + (size_t)(lane & 15) * ld + ((lane >> 4) << 4);
  v16u f;
#pragma unroll
  for (int i = 0; i < 8; ++i) { f[i] = p[i]; f[i + 8] = p[i + 8]; }
  return as_bf(f);
}
// A with only nrows valid rows (rest zero) — pads batch(8) to M=16
__device__ __forceinline__ v16bf load_a_bf16_rows(const bf16_t* S, int ld, int nrows) {
  int lane = threadIdx.x & 31;
  int r = lane & 15;
  v16u f;
  if (r < nrows) {
    const bf16_t* p = S + (size_t)r * ld + ((lane >> 4) << 4);
#pragma unroll
    for (int i = 0; i < 8; ++i) { f[i] = p[i]; f[i + 8] = p[i + 8]; }
  } else {
#pragma unroll
    for (int i = 0; i < 16; ++i) f[i] = 0;
  }
  return as_bf(f);
}
// B: 32x16 KxN from transposed source: B[k][n] = S[n*ld + k] -> 16 contiguous bf16 per lane
__device__ __forceinline__ v16bf load_b_trans_bf16(const bf16_t* S, int ld) {
  int lane = threadIdx.x & 31;
  const bf16_t* p = S + (size_t)(lane & 15) * ld + ((lane >> 4) << 4);
  v16u f;
#pragma unroll
  for (int j = 0; j < 16; ++j) f[j] = p[j];
  return as_bf(f);
}
__device__ __forceinline__ v16bf load_b_trans_f32(const float* S, int ld) {
  int lane = threadIdx.x & 31;
  const float* p = S + (size_t)(lane & 15) * ld + ((lane >> 4) << 4);
  v16u f;
#pragma unroll
  for (int j = 0; j < 16; ++j) f[j] = f2bf(p[j]);
  return as_bf(f);
}
__device__ __forceinline__ v8f wmma_bf16(v16bf a, v16bf b, v8f c) {
  return __builtin_amdgcn_wmma_f32_16x16x32_bf16(false, a, false, b, (short)0, c, false, false);
}

// ------- transpose + convert: out[C x R] bf16 = (in[R x C])^T, batched over grid.z -------
// LDS-tiled: coalesced global reads AND writes.
__global__ void k_cvt_t(const float* __restrict__ in, bf16_t* __restrict__ out, int R, int C) {
  __shared__ bf16_t tile[64][65];
  const float* src = in + (size_t)blockIdx.z * R * C;
  bf16_t* dst = out + (size_t)blockIdx.z * R * C;
  int r0 = blockIdx.y * 64, c0 = blockIdx.x * 64;
  int l = threadIdx.x;
#pragma unroll
  for (int i = 0; i < 16; ++i) {
    int e = l + i * 256;
    int r = e >> 6, c = e & 63;
    tile[r][c] = f2bf(src[(size_t)(r0 + r) * C + c0 + c]);
  }
  __syncthreads();
#pragma unroll
  for (int i = 0; i < 16; ++i) {
    int e = l + i * 256;
    int r = e >> 6, c = e & 63;                 // coords in transposed tile
    dst[(size_t)(c0 + r) * R + r0 + c] = tile[c][r];
  }
}

// ---------------- rmsnorm: one block (128 thr) per 512-wide row ----------------
__global__ void k_rmsnorm(const float* __restrict__ in, const float* __restrict__ w,
                          float* __restrict__ outf, bf16_t* __restrict__ outb) {
  __shared__ float red[128];
  size_t row = blockIdx.x;
  const float* p = in + row * D_MODEL;
  int t = threadIdx.x;
  float x0 = p[t], x1 = p[t + 128], x2 = p[t + 256], x3 = p[t + 384];
  red[t] = x0 * x0 + x1 * x1 + x2 * x2 + x3 * x3;
  __syncthreads();
  for (int s = 64; s > 0; s >>= 1) {
    if (t < s) red[t] += red[t + s];
    __syncthreads();
  }
  float rs = rsqrtf(red[0] * (1.0f / (float)D_MODEL) + NORM_EPS);
  float y0 = x0 * rs * w[t], y1 = x1 * rs * w[t + 128];
  float y2 = x2 * rs * w[t + 256], y3 = x3 * rs * w[t + 384];
  float* of = outf + row * D_MODEL;
  bf16_t* ob = outb + row * D_MODEL;
  of[t] = y0; of[t + 128] = y1; of[t + 256] = y2; of[t + 384] = y3;
  ob[t] = f2bf(y0); ob[t + 128] = f2bf(y1); ob[t + 256] = f2bf(y2); ob[t + 384] = f2bf(y3);
}

// ---------------- QKV: per (row-tile, head, mat) GEMM [16x512]x[512x64] ----------------
// Weights pre-transposed: WT is [64 x 512] per head. V is stored TRANSPOSED: [H][B][64][512].
__global__ void k_qkv(const bf16_t* __restrict__ xb,
                      const bf16_t* __restrict__ WqT, const float* __restrict__ bq,
                      const bf16_t* __restrict__ WkT, const float* __restrict__ bk,
                      const bf16_t* __restrict__ WvT, const float* __restrict__ bv,
                      bf16_t* __restrict__ qb, bf16_t* __restrict__ kb, bf16_t* __restrict__ vbT) {
  int rt = blockIdx.x, h = blockIdx.y, m = blockIdx.z;
  const bf16_t* WT  = (m == 0 ? WqT : (m == 1 ? WkT : WvT)) + (size_t)h * HEAD_SIZE * D_MODEL;
  const float* bias = (m == 0 ? bq : (m == 1 ? bk : bv)) + h * HEAD_SIZE;
  int wv = threadIdx.x >> 5, lane = threadIdx.x & 31;
  int n0 = wv * 16;
  v8f acc = {};
#pragma unroll
  for (int kk = 0; kk < D_MODEL; kk += 32) {
    v16bf a = load_a_bf16(xb + (size_t)(rt * 16) * D_MODEL + kk, D_MODEL);
    v16bf b = load_b_trans_bf16(WT + (size_t)n0 * D_MODEL + kk, D_MODEL);
    acc = wmma_bf16(a, b, acc);
  }
  int n = lane & 15, mb = (lane >> 4) * 8;
  float bb = bias[n0 + n];
  if (m == 2) {  // store v transposed: vT[((h*B + b)*64 + c)*SEQ + t]
    int b_ = (rt * 16) >> 9;
    int tb = (rt * 16) & (SEQ - 1);
#pragma unroll
    for (int i = 0; i < 8; ++i)
      vbT[(((size_t)h * BATCH + b_) * HEAD_SIZE + n0 + n) * SEQ + tb + mb + i] = f2bf(acc[i] + bb);
  } else {
    bf16_t* outp = (m == 0 ? qb : kb) + ((size_t)h * ROWS + rt * 16) * HEAD_SIZE;
#pragma unroll
    for (int i = 0; i < 8; ++i)
      outp[(size_t)(mb + i) * HEAD_SIZE + n0 + n] = f2bf(acc[i] + bb);
  }
}

// ---------------- rel bias: per (h,t) GEMM, M = batch(8, padded 16), K=64, N=512 ----------------
// rel_pos (512 MB) is streamed exactly once, fully coalesced; result -> relbuf f32.
__global__ void k_rel(const bf16_t* __restrict__ qb, const float* __restrict__ rel_pos,
                      float* __restrict__ relbuf) {
  int t = blockIdx.x, h = blockIdx.y;
  int wv = threadIdx.x >> 5, lane = threadIdx.x & 31;
  const bf16_t* qbase = qb + ((size_t)h * ROWS + t) * HEAD_SIZE;  // b-stride = SEQ*HEAD_SIZE
  v16bf a0 = load_a_bf16_rows(qbase,      SEQ * HEAD_SIZE, BATCH);
  v16bf a1 = load_a_bf16_rows(qbase + 32, SEQ * HEAD_SIZE, BATCH);
  const float* R = rel_pos + ((size_t)h * SEQ + t) * SEQ * HEAD_SIZE;
#pragma unroll
  for (int j = 0; j < 4; ++j) {
    int v0 = wv * 64 + j * 16;
    const float* Rt = R + (size_t)v0 * HEAD_SIZE;
    __builtin_prefetch(Rt + 16 * HEAD_SIZE, 0, 1);   // stream next tile (global_prefetch_b8)
    v8f acc = {};
    acc = wmma_bf16(a0, load_b_trans_f32(Rt,      HEAD_SIZE), acc);
    acc = wmma_bf16(a1, load_b_trans_f32(Rt + 32, HEAD_SIZE), acc);
    int n = lane & 15, mb = (lane >> 4) * 8;
    if (mb == 0) {  // rows 0..7 = valid batches, live in lanes 0..15
#pragma unroll
      for (int i = 0; i < 8; ++i)
        relbuf[(((size_t)i * N_HEAD + h) * SEQ + t) * SEQ + v0 + n] = acc[i];
    }
  }
}

// ---------------- scores: q k^T * sqrt(hs) + rel, in-place into relbuf ----------------
__global__ void k_scores(const bf16_t* __restrict__ qb, const bf16_t* __restrict__ kb,
                         float* __restrict__ relbuf) {
  int tt = blockIdx.x, h = blockIdx.y, b = blockIdx.z;
  int wv = threadIdx.x >> 5, lane = threadIdx.x & 31;
  const bf16_t* Q = qb + ((size_t)h * ROWS + b * SEQ + tt * 16) * HEAD_SIZE;
  v16bf a0 = load_a_bf16(Q,      HEAD_SIZE);
  v16bf a1 = load_a_bf16(Q + 32, HEAD_SIZE);
  const float scale = 8.0f;  // faithful quirk: * sqrt(head_size)
  float* L = relbuf + (((size_t)b * N_HEAD + h) * SEQ + tt * 16) * SEQ;
#pragma unroll
  for (int j = 0; j < 4; ++j) {
    int v0 = wv * 64 + j * 16;
    const bf16_t* K = kb + ((size_t)h * ROWS + b * SEQ + v0) * HEAD_SIZE;
    v8f acc = {};
    acc = wmma_bf16(a0, load_b_trans_bf16(K,      HEAD_SIZE), acc);
    acc = wmma_bf16(a1, load_b_trans_bf16(K + 32, HEAD_SIZE), acc);
    int n = lane & 15, mb = (lane >> 4) * 8;
#pragma unroll
    for (int i = 0; i < 8; ++i) {
      size_t idx = (size_t)(mb + i) * SEQ + v0 + n;
      L[idx] = acc[i] * scale + L[idx];
    }
  }
}

// ---------------- softmax: one wave32 per 512-wide row, shuffle reductions ----------------
__global__ void k_softmax(const float* __restrict__ logits, bf16_t* __restrict__ att) {
  int wv = threadIdx.x >> 5, lane = threadIdx.x & 31;
  size_t row = (size_t)blockIdx.x * 8 + wv;
  const float* p = logits + row * SEQ;
  float v[16];
  float mx = -3.4e38f;
#pragma unroll
  for (int i = 0; i < 16; ++i) { v[i] = p[i * 32 + lane]; mx = fmaxf(mx, v[i]); }
#pragma unroll
  for (int s = 16; s > 0; s >>= 1) mx = fmaxf(mx, __shfl_xor(mx, s, 32));
  float sum = 0.f;
#pragma unroll
  for (int i = 0; i < 16; ++i) { v[i] = __expf(v[i] - mx); sum += v[i]; }
#pragma unroll
  for (int s = 16; s > 0; s >>= 1) sum += __shfl_xor(sum, s, 32);
  float inv = 1.0f / sum;
  bf16_t* o = att + row * SEQ;
#pragma unroll
  for (int i = 0; i < 16; ++i) o[i * 32 + lane] = f2bf(v[i] * inv);
}

// ---------------- att @ V: per (b,h,t-tile) GEMM [16x512]x[512x64], V^T layout ----------------
__global__ void k_av(const bf16_t* __restrict__ att, const bf16_t* __restrict__ vbT,
                     bf16_t* __restrict__ attout) {
  int tt = blockIdx.x, h = blockIdx.y, b = blockIdx.z;
  int wv = threadIdx.x >> 5, lane = threadIdx.x & 31;
  int n0 = wv * 16;
  const bf16_t* A  = att + (((size_t)b * N_HEAD + h) * SEQ + tt * 16) * SEQ;
  const bf16_t* VT = vbT + ((size_t)h * BATCH + b) * HEAD_SIZE * SEQ;
  v8f acc = {};
#pragma unroll
  for (int kk = 0; kk < SEQ; kk += 32) {
    v16bf a  = load_a_bf16(A + kk, SEQ);
    v16bf bb = load_b_trans_bf16(VT + (size_t)n0 * SEQ + kk, SEQ);
    acc = wmma_bf16(a, bb, acc);
  }
  int n = lane & 15, mb = (lane >> 4) * 8;
#pragma unroll
  for (int i = 0; i < 8; ++i)  // concat heads: [row][h*64 + c]
    attout[((size_t)b * SEQ + tt * 16 + mb + i) * D_MODEL + h * HEAD_SIZE + n0 + n] = f2bf(acc[i]);
}

// ---------------- proj + bias + residual (W^T layout) ----------------
__global__ void k_proj(const bf16_t* __restrict__ attout, const bf16_t* __restrict__ WpT,
                       const float* __restrict__ bp, const float* __restrict__ x,
                       float* __restrict__ src2) {
  int rt = blockIdx.x, cc = blockIdx.y;
  int wv = threadIdx.x >> 5, lane = threadIdx.x & 31;
  int col0 = cc * 64 + wv * 16;
  v8f acc = {};
#pragma unroll
  for (int kk = 0; kk < D_MODEL; kk += 32) {
    v16bf a  = load_a_bf16(attout + (size_t)(rt * 16) * D_MODEL + kk, D_MODEL);
    v16bf bb = load_b_trans_bf16(WpT + (size_t)col0 * D_MODEL + kk, D_MODEL);
    acc = wmma_bf16(a, bb, acc);
  }
  int n = lane & 15, mb = (lane >> 4) * 8;
  float bias = bp[col0 + n];
#pragma unroll
  for (int i = 0; i < 8; ++i) {
    size_t idx = (size_t)(rt * 16 + mb + i) * D_MODEL + col0 + n;
    src2[idx] = acc[i] + bias + x[idx];
  }
}

// ---------------- FF1 + exact GELU (W^T layout) ----------------
__global__ void k_ff1(const bf16_t* __restrict__ yb, const bf16_t* __restrict__ W1T,
                      const float* __restrict__ b1, bf16_t* __restrict__ hb) {
  int rt = blockIdx.x, cc = blockIdx.y;
  int wv = threadIdx.x >> 5, lane = threadIdx.x & 31;
  int col0 = cc * 64 + wv * 16;
  v8f acc = {};
#pragma unroll
  for (int kk = 0; kk < D_MODEL; kk += 32) {
    v16bf a  = load_a_bf16(yb + (size_t)(rt * 16) * D_MODEL + kk, D_MODEL);
    v16bf bb = load_b_trans_bf16(W1T + (size_t)col0 * D_MODEL + kk, D_MODEL);
    acc = wmma_bf16(a, bb, acc);
  }
  int n = lane & 15, mb = (lane >> 4) * 8;
  float bias = b1[col0 + n];
#pragma unroll
  for (int i = 0; i < 8; ++i) {
    float u = acc[i] + bias;
    float g = 0.5f * u * (1.0f + erff(u * 0.70710678118f));
    hb[(size_t)(rt * 16 + mb + i) * FF_DIM + col0 + n] = f2bf(g);
  }
}

// ---------------- FF2 + bias + residual -> d_out (W^T layout) ----------------
__global__ void k_ff2(const bf16_t* __restrict__ hb, const bf16_t* __restrict__ W2T,
                      const float* __restrict__ b2, const float* __restrict__ y,
                      float* __restrict__ out) {
  int rt = blockIdx.x, cc = blockIdx.y;
  int wv = threadIdx.x >> 5, lane = threadIdx.x & 31;
  int col0 = cc * 64 + wv * 16;
  v8f acc = {};
#pragma unroll
  for (int kk = 0; kk < FF_DIM; kk += 32) {
    v16bf a  = load_a_bf16(hb + (size_t)(rt * 16) * FF_DIM + kk, FF_DIM);
    v16bf bb = load_b_trans_bf16(W2T + (size_t)col0 * FF_DIM + kk, FF_DIM);
    acc = wmma_bf16(a, bb, acc);
  }
  int n = lane & 15, mb = (lane >> 4) * 8;
  float bias = b2[col0 + n];
#pragma unroll
  for (int i = 0; i < 8; ++i) {
    size_t idx = (size_t)(rt * 16 + mb + i) * D_MODEL + col0 + n;
    out[idx] = acc[i] + bias + y[idx];
  }
}

extern "C" void kernel_launch(void* const* d_in, const int* in_sizes, int n_in,
                              void* d_out, int out_size, void* d_ws, size_t ws_size,
                              hipStream_t stream) {
  const float* src    = (const float*)d_in[0];
  const float* norm_w = (const float*)d_in[1];
  const float* Wq = (const float*)d_in[2];  const float* bq = (const float*)d_in[3];
  const float* Wk = (const float*)d_in[4];  const float* bk = (const float*)d_in[5];
  const float* Wv = (const float*)d_in[6];  const float* bv = (const float*)d_in[7];
  const float* rel_pos = (const float*)d_in[8];
  const float* pw = (const float*)d_in[9];  const float* pb = (const float*)d_in[10];
  const float* w1 = (const float*)d_in[11]; const float* b1 = (const float*)d_in[12];
  const float* w2 = (const float*)d_in[13]; const float* b2 = (const float*)d_in[14];

  char* ws = (char*)d_ws;
  float*  xf     = (float*) (ws + ((size_t)0   << 20));  //  8 MB  normed x (f32, residual)
  bf16_t* xb     = (bf16_t*)(ws + ((size_t)8   << 20));  //  4 MB  normed x (bf16)
  bf16_t* qb     = (bf16_t*)(ws + ((size_t)12  << 20));  //  4 MB  q [H][B*T][64]
  bf16_t* kbuf   = (bf16_t*)(ws + ((size_t)16  << 20));  //  4 MB  k [H][B*T][64]
  bf16_t* vbT    = (bf16_t*)(ws + ((size_t)20  << 20));  //  4 MB  v^T [H][B][64][512]
  float*  relbuf = (float*) (ws + ((size_t)24  << 20));  // 64 MB  rel bias -> logits (in-place)
  bf16_t* att    = (bf16_t*)(ws + ((size_t)88  << 20));  // 32 MB  softmax(att)
  bf16_t* attout = (bf16_t*)(ws + ((size_t)120 << 20));  //  4 MB  head-concat output
  float*  src2   = (float*) (ws + ((size_t)124 << 20));  //  8 MB  x + att_out
  float*  yf     = (float*) (ws + ((size_t)132 << 20));  //  8 MB  y (f32, residual)
  bf16_t* yb     = (bf16_t*)(ws + ((size_t)140 << 20));  //  4 MB  y (bf16)
  bf16_t* hb     = (bf16_t*)(ws + ((size_t)144 << 20));  // 16 MB  gelu hidden
  bf16_t* wqT    = (bf16_t*)(ws + ((size_t)160 << 20));  // 512 KB bf16 W^T per head [64][512]
  bf16_t* wkT    = (bf16_t*)(ws + ((size_t)161 << 20));
  bf16_t* wvT    = (bf16_t*)(ws + ((size_t)162 << 20));
  bf16_t* pwT    = (bf16_t*)(ws + ((size_t)163 << 20));  // [512][512]
  bf16_t* w1T    = (bf16_t*)(ws + ((size_t)164 << 20));  //  2 MB  [2048][512]
  bf16_t* w2T    = (bf16_t*)(ws + ((size_t)166 << 20));  //  2 MB  [512][2048]

  // one-time weight transpose + bf16 convert (LDS-tiled, coalesced both ways)
  k_cvt_t<<<dim3(1, 8, N_HEAD), 256, 0, stream>>>(Wq, wqT, D_MODEL, HEAD_SIZE);
  k_cvt_t<<<dim3(1, 8, N_HEAD), 256, 0, stream>>>(Wk, wkT, D_MODEL, HEAD_SIZE);
  k_cvt_t<<<dim3(1, 8, N_HEAD), 256, 0, stream>>>(Wv, wvT, D_MODEL, HEAD_SIZE);
  k_cvt_t<<<dim3(8, 8, 1),      256, 0, stream>>>(pw, pwT, D_MODEL, D_MODEL);
  k_cvt_t<<<dim3(32, 8, 1),     256, 0, stream>>>(w1, w1T, D_MODEL, FF_DIM);
  k_cvt_t<<<dim3(8, 32, 1),     256, 0, stream>>>(w2, w2T, FF_DIM, D_MODEL);

  k_rmsnorm<<<ROWS, 128, 0, stream>>>(src, norm_w, xf, xb);
  k_qkv    <<<dim3(ROWS / 16, N_HEAD, 3), 128, 0, stream>>>(xb, wqT, bq, wkT, bk, wvT, bv,
                                                            qb, kbuf, vbT);
  k_rel    <<<dim3(SEQ, N_HEAD), 256, 0, stream>>>(qb, rel_pos, relbuf);
  k_scores <<<dim3(SEQ / 16, N_HEAD, BATCH), 256, 0, stream>>>(qb, kbuf, relbuf);
  k_softmax<<<(BATCH * N_HEAD * SEQ) / 8, 256, 0, stream>>>(relbuf, att);
  k_av     <<<dim3(SEQ / 16, N_HEAD, BATCH), 128, 0, stream>>>(att, vbT, attout);
  k_proj   <<<dim3(ROWS / 16, D_MODEL / 64), 128, 0, stream>>>(attout, pwT, pb, xf, src2);
  k_rmsnorm<<<ROWS, 128, 0, stream>>>(src2, norm_w, yf, yb);
  k_ff1    <<<dim3(ROWS / 16, FF_DIM / 64), 128, 0, stream>>>(yb, w1T, b1, hb);
  k_ff2    <<<dim3(ROWS / 16, D_MODEL / 64), 128, 0, stream>>>(hb, w2T, b2, yf, (float*)d_out);
}
